// CharVanilla_86990267613901
// MI455X (gfx1250) — matrix operand order repaired and checked by reading, not compiled, not run on hardware
//
#include <hip/hip_runtime.h>
#include <hip/hip_bf16.h>
#include <stdint.h>

typedef _Float16 v16h __attribute__((ext_vector_type(16)));
typedef __fp16   v2fp __attribute__((ext_vector_type(2)));
typedef float    v8f  __attribute__((ext_vector_type(8)));
typedef float    v4f  __attribute__((ext_vector_type(4)));
typedef unsigned int v8u __attribute__((ext_vector_type(8)));

#define VOCAB 256
#define HID   20
#define NLAB  15
#define TLEN  512
#define BATCH 8192
#define ROWS_PER_WAVE   16
#define WAVES_PER_BLOCK 4
#define THREADS (WAVES_PER_BLOCK * 32)

#define SWAPX16 0x401F   // ds_swizzle group-of-32: xor=0x10, or=0, and=0x1f

union BU { v16h h; v8u u; };

__device__ __forceinline__ float fast_tanh(float x) {
#if __has_builtin(__builtin_amdgcn_tanhf)
    return __builtin_amdgcn_tanhf(x);                  // v_tanh_f32
#elif __has_builtin(__builtin_amdgcn_tanh_f32)
    return __builtin_amdgcn_tanh_f32(x);
#else
    // branchless: tanh(x) = 1 - 2/(exp(2x)+1); saturates correctly at +-1
    float e = __expf(2.0f * x);                        // v_mul + v_exp_f32
    return 1.0f - 2.0f * __builtin_amdgcn_rcpf(e + 1.0f);
#endif
}

__device__ __forceinline__ unsigned int pack2(float a, float b) {
#if __has_builtin(__builtin_amdgcn_cvt_pkrtz)
    union { v2fp h; unsigned int u; } c;
    c.h = __builtin_amdgcn_cvt_pkrtz(a, b);            // v_cvt_pk_rtz_f16_f32
    return c.u;
#else
    union { _Float16 h[2]; unsigned int u; } c;
    c.h[0] = (_Float16)a; c.h[1] = (_Float16)b; return c.u;
#endif
}

__global__ __launch_bounds__(THREADS)
void CharVanilla_rnn_wmma(const float* __restrict__ Wx,
                          const float* __restrict__ Wh,
                          const float* __restrict__ b,
                          const float* __restrict__ Wd,
                          const float* __restrict__ bd,
                          const int*   __restrict__ x,
                          float*       __restrict__ out)
{
    __shared__ float WxL[VOCAB * HID];                    // Wx + b, 20 KB
    __shared__ float WdL[HID * NLAB];
    __shared__ float bdL[NLAB];
    __shared__ float h32[WAVES_PER_BLOCK][16][HID + 1];   // final h for dense

    const int tid = threadIdx.x;
    for (int i = tid; i < VOCAB * HID; i += THREADS)
        WxL[i] = Wx[i] + b[i % HID];
    for (int i = tid; i < HID * NLAB; i += THREADS)
        WdL[i] = Wd[i];
    if (tid < NLAB) bdL[tid] = bd[tid];
    __syncthreads();

    const int  lane  = tid & 31;
    const int  wslot = tid >> 5;
    const int  hl    = lane >> 4;        // half-wave id
    const int  n0    = lane & 15;        // batch column within tile
    const long base  = (long)(blockIdx.x * WAVES_PER_BLOCK + wslot) * ROWS_PER_WAVE;

    // Transposed formulation: D = Wh^T x h^T + xe^T.
    // A = Wh^T in 16x32 f16 A layout (constant), two M tiles: hid_out 0..15, 16..19.
    // A layout: lane M = lane&15 (tile a) / 16+(lane&15) (tile b);
    // element e -> K = e + 16*hl, zero-padded for K >= 20.
    v16h Aa = {}, Ab = {};
    {
        const int m0 = n0, m1 = 16 + n0;
        #pragma unroll
        for (int e = 0; e < 16; ++e) {
            int K = e + 16 * hl;
            if (K < HID) {
                Aa[e] = (_Float16)Wh[K * HID + m0];
                Ab[e] = (m1 < HID) ? (_Float16)Wh[K * HID + m1] : (_Float16)0.f;
            }
        }
    }

    // B = h^T in 32x16 f16 B layout: lane-half hl holds K = 16*hl + e, N = lane&15.
    BU B; B.h = (v16h){};                 // h0 = 0

    float tha[8];                         // tanh(D_a): hid (r + 8*hl) of batch n0
    float thb[4];                         // tanh(D_b): hid (16+r) (valid on hl==0)
    #pragma unroll
    for (int r = 0; r < 8; ++r) tha[r] = 0.f;
    #pragma unroll
    for (int r = 0; r < 4; ++r) thb[r] = 0.f;

    const int* xrow = x + (base + n0) * TLEN;   // this lane's token row

    for (int t0 = 0; t0 < TLEN; t0 += 4) {
        if ((t0 & 127) == 0 && t0 + 128 < TLEN)
            __builtin_prefetch(xrow + t0 + 128, 0, 1);     // global_prefetch_b8

        const int4 tq = *(const int4*)(xrow + t0);          // 4 tokens, one b128
        int toks[4] = { tq.x, tq.y, tq.z, tq.w };

        #pragma unroll
        for (int s = 0; s < 4; ++s) {
            // C = xe^T: lane needs hid components (r + 8*hl) and (16 + r + 8*hl)
            // of its own token's embedding row -> contiguous ds_load_b128 x3.
            const float* we = &WxL[toks[s] * HID + 8 * hl];
            v4f ca_lo = *(const v4f*)(we);
            v4f ca_hi = *(const v4f*)(we + 4);
            v4f cb4   = *(const v4f*)(we + 16);   // garbage on hl==1, discarded

            v8f Ca, Cb;
            #pragma unroll
            for (int r = 0; r < 4; ++r) {
                Ca[r]     = ca_lo[r];
                Ca[4 + r] = ca_hi[r];
                Cb[r]     = cb4[r];
                Cb[4 + r] = 0.f;
            }

            v8f Da = __builtin_amdgcn_wmma_f32_16x16x32_f16(false, Aa, false, B.h,
                                                            (short)0, Ca, false, false);
            v8f Db = __builtin_amdgcn_wmma_f32_16x16x32_f16(false, Ab, false, B.h,
                                                            (short)0, Cb, false, false);

            #pragma unroll
            for (int r = 0; r < 8; ++r) tha[r] = fast_tanh(Da[r]);
            #pragma unroll
            for (int r = 0; r < 4; ++r) thb[r] = fast_tanh(Db[r]);

            // Rebuild B (h^T) branchlessly:
            //  hl==0 lanes: e0..7 = local tha, e8..15 = partner's tha (swizzle)
            //  hl==1 lanes: e0..3 = partner's thb (swizzle), rest 0
            unsigned int pa[4], sa[4], pb[2], sb[2];
            #pragma unroll
            for (int j = 0; j < 4; ++j) pa[j] = pack2(tha[2*j], tha[2*j + 1]);
            pb[0] = pack2(thb[0], thb[1]);
            pb[1] = pack2(thb[2], thb[3]);
            #pragma unroll
            for (int j = 0; j < 4; ++j)
                sa[j] = __builtin_amdgcn_ds_swizzle(pa[j], SWAPX16);
            sb[0] = __builtin_amdgcn_ds_swizzle(pb[0], SWAPX16);
            sb[1] = __builtin_amdgcn_ds_swizzle(pb[1], SWAPX16);

            BU Bn;
            Bn.u[0] = hl ? sb[0] : pa[0];
            Bn.u[1] = hl ? sb[1] : pa[1];
            Bn.u[2] = hl ? 0u    : pa[2];
            Bn.u[3] = hl ? 0u    : pa[3];
            Bn.u[4] = hl ? 0u    : sa[0];
            Bn.u[5] = hl ? 0u    : sa[1];
            Bn.u[6] = hl ? 0u    : sa[2];
            Bn.u[7] = hl ? 0u    : sa[3];
            B = Bn;
        }
    }

    // Final hidden state (f32) -> LDS tile for the dense head.
    if (hl == 0) {
        #pragma unroll
        for (int r = 0; r < 8; ++r) h32[wslot][n0][r] = tha[r];
        #pragma unroll
        for (int r = 0; r < 4; ++r) h32[wslot][n0][16 + r] = thb[r];
    } else {
        #pragma unroll
        for (int r = 0; r < 8; ++r) h32[wslot][n0][8 + r] = tha[r];
    }
    __builtin_amdgcn_wave_barrier();   // same-wave DS in-order; fence code motion

    // Dense (20 -> 15) + softmax, one batch row per lane (lanes 0..15).
    if (lane < 16) {
        const int m = lane;
        float hv[HID];
        #pragma unroll
        for (int k = 0; k < HID; ++k) hv[k] = h32[wslot][m][k];

        float logits[NLAB];
        float mx = -1e30f;
        #pragma unroll
        for (int j = 0; j < NLAB; ++j) {
            float sacc = bdL[j];
            #pragma unroll
            for (int k = 0; k < HID; ++k) sacc += hv[k] * WdL[k * NLAB + j];
            logits[j] = sacc;
            mx = fmaxf(mx, sacc);
        }
        float den = 0.f;
        #pragma unroll
        for (int j = 0; j < NLAB; ++j) {
            float e = __expf(logits[j] - mx);
            logits[j] = e;
            den += e;
        }
        float inv = 1.f / den;
        float* orow = out + (base + m) * NLAB;
        #pragma unroll
        for (int j = 0; j < NLAB; ++j) orow[j] = logits[j] * inv;
    }
}

extern "C" void kernel_launch(void* const* d_in, const int* in_sizes, int n_in,
                              void* d_out, int out_size, void* d_ws, size_t ws_size,
                              hipStream_t stream) {
    (void)in_sizes; (void)n_in; (void)out_size; (void)d_ws; (void)ws_size;
    const float* Wx = (const float*)d_in[0];
    const float* Wh = (const float*)d_in[1];
    const float* b  = (const float*)d_in[2];
    const float* Wd = (const float*)d_in[3];
    const float* bd = (const float*)d_in[4];
    const int*   x  = (const int*)d_in[5];
    // d_in[6] = drop_rate (0) -> no-op per reference semantics
    float* out = (float*)d_out;

    dim3 grid(BATCH / (ROWS_PER_WAVE * WAVES_PER_BLOCK));   // 128 blocks
    CharVanilla_rnn_wmma<<<grid, THREADS, 0, stream>>>(Wx, Wh, b, Wd, bd, x, out);
}
